// MultiHeadAttention_5085241279147
// MI455X (gfx1250) — matrix operand order repaired and checked by reading, not compiled, not run on hardware
//
#include <hip/hip_runtime.h>
#include <hip/hip_bf16.h>
#include <cstdint>

typedef __attribute__((ext_vector_type(16))) _Float16 v16h;
typedef __attribute__((ext_vector_type(8)))  _Float16 v8h;
typedef __attribute__((ext_vector_type(8)))  float    v8f;

#define D_MODEL 1024
#define N_HEAD  16
#define DPH     64
#define BATCH   4
#define SEQ     2048
#define M_TOT   (BATCH*SEQ)          // 8192 rows
#define SP      2052                 // LDS row stride (f32) to spread banks
// S + AVG + red(256) + rmax(16) + rsum(16) + PP(1024) + mask(16*513 u32)
#define SMEM_FLOATS (2*16*SP + 256 + 16 + 16 + 4*256 + 16*513)
#define SMEM_BYTES  (SMEM_FLOATS * sizeof(float))   // 300,736 B <= 320KB

// ---------------------------------------------------------------------------
// WMMA fragment loaders (CDNA5 16x16x32 f16, wave32)
// ---------------------------------------------------------------------------

// A fragment: 16x32 f16 from row-major [M][K] (ld multiple of 8, k0 mult of 32)
// lane<16: M=lane, K={k0+0..7, k0+16..23}; lane>=16: M=lane-16, K={+8..15,+24..31}
__device__ __forceinline__ v16h load_a(const _Float16* __restrict__ base, int ld, int k0) {
  const int lane = threadIdx.x & 31;
  const int m    = lane & 15;
  const int klo  = (lane >> 4) << 3;          // 0 or 8
  const _Float16* p = base + (size_t)m * ld + k0 + klo;
  union { v16h v; v8h h[2]; } u;
  u.h[0] = *(const v8h*)(p);
  u.h[1] = *(const v8h*)(p + 16);
  return u.v;
}

// B fragment: 32x16 f16 supplied as B^T row-major [N][K].
// lane&15 = N column; lane<16: K={k0+0..7, k0+16..23}; lane>=16: K={+8..15,+24..31}.
__device__ __forceinline__ v16h load_bt(const _Float16* __restrict__ base, int ld, int k0) {
  const int lane = threadIdx.x & 31;
  const int n    = lane & 15;
  const int kb   = k0 + ((lane >> 4) << 3);   // +0 or +8
  const _Float16* p = base + (size_t)n * ld + kb;
  union { v16h v; v8h h[2]; } u;
  u.h[0] = *(const v8h*)(p);
  u.h[1] = *(const v8h*)(p + 16);
  return u.v;
}

// A fragment from f32 LDS (softmax probs), converting to f16 on the fly.
__device__ __forceinline__ v16h load_a_lds(const float* __restrict__ base, int ld, int k0) {
  const int lane = threadIdx.x & 31;
  const int m    = lane & 15;
  const int klo  = (lane >> 4) << 3;
  const float* p = base + m * ld + k0 + klo;
  v16h v;
#pragma unroll
  for (int j = 0; j < 8; ++j) {
    v[j]     = (_Float16)p[j];
    v[j + 8] = (_Float16)p[j + 16];
  }
  return v;
}

__device__ __forceinline__ v8f wmma16(v16h a, v16h b, v8f c) {
  return __builtin_amdgcn_wmma_f32_16x16x32_f16(false, a, false, b, (short)0, c, false, false);
}

// ---------------------------------------------------------------------------
// Kernel 0: f32 -> f16 convert
// ---------------------------------------------------------------------------
__global__ void mha_cvt_f16(const float* __restrict__ src, _Float16* __restrict__ dst, int n) {
  int i = blockIdx.x * blockDim.x + threadIdx.x;
  if (i < n) dst[i] = (_Float16)src[i];
}

// ---------------------------------------------------------------------------
// Kernel 1: fused QKV projection.  P = X @ W^T + b, head-split outputs.
// grid = (512, 48), block = 128 (4 waves); wave computes one 16x16 tile.
// ---------------------------------------------------------------------------
__global__ void mha_qkv_proj(const _Float16* __restrict__ xk,
                             const _Float16* __restrict__ xq,
                             const _Float16* __restrict__ xv,
                             const _Float16* __restrict__ w,     // [3072][1024] = B^T rows
                             const float*    __restrict__ bias,  // [3072] f32
                             _Float16* __restrict__ qh,          // [B][H][L][64]
                             _Float16* __restrict__ kh,          // [B][H][L][64]
                             _Float16* __restrict__ vT) {        // [B][H][64][L]
  const int wave = threadIdx.x >> 5;
  const int lane = threadIdx.x & 31;
  const int m0   = blockIdx.x * 16;
  const int col0 = (blockIdx.y * 4 + wave) * 16;   // global output channel [0,3072)
  const int mat  = col0 >> 10;                     // 0=k, 1=q, 2=v

  const _Float16* x = (mat == 0) ? xk : (mat == 1) ? xq : xv;
  const _Float16* abase = x + (size_t)m0 * D_MODEL;
  const _Float16* bbase = w + (size_t)col0 * D_MODEL;

  v8f acc = {0.f,0.f,0.f,0.f,0.f,0.f,0.f,0.f};
  for (int k0 = 0; k0 < D_MODEL; k0 += 32) {
    acc = wmma16(load_a(abase, D_MODEL, k0), load_bt(bbase, D_MODEL, k0), acc);
  }

  const int n  = lane & 15;
  const int cg = col0 + n;
  const int c  = cg & (D_MODEL - 1);
  const int h  = c >> 6;
  const int d  = c & 63;
  const float bv = bias[cg];
  const int mb = (lane >> 4) << 3;
#pragma unroll
  for (int i = 0; i < 8; ++i) {
    const int Mg = m0 + mb + i;
    const int bb = Mg >> 11;
    const int l  = Mg & (SEQ - 1);
    const float val = acc[i] + bv;
    if (mat == 0) {
      kh[(((size_t)(bb * N_HEAD + h)) * SEQ + l) * DPH + d] = (_Float16)val;
    } else if (mat == 1) {
      qh[(((size_t)(bb * N_HEAD + h)) * SEQ + l) * DPH + d] = (_Float16)(val * 0.125f);
    } else {
      vT[(((size_t)(bb * N_HEAD + h)) * DPH + d) * SEQ + l] = (_Float16)val;
    }
  }
}

// ---------------------------------------------------------------------------
// Kernel 2: fused attention.  One block per (batch, 16-query tile); loops all
// heads so attn_avg accumulates in LDS.  block = 256 (8 waves, 2/SIMD).
// attn_mask staged ONCE per block into LDS bytes (reused by all 16 heads).
// ---------------------------------------------------------------------------
__global__ void mha_attn(const _Float16* __restrict__ qh,
                         const _Float16* __restrict__ kh,
                         const _Float16* __restrict__ vT,
                         const int*      __restrict__ mask,      // [B][L][L]
                         const float*    __restrict__ qmask,     // [B][L]
                         _Float16* __restrict__ ctx,             // [B*L][1024] f16
                         float*    __restrict__ avg_out) {       // [B][L][L]
  extern __shared__ char smem_raw[];
  float* S    = (float*)smem_raw;        // [16][SP] scores -> probs
  float* AVG  = S + 16 * SP;             // [16][SP] running head-average
  float* red  = AVG + 16 * SP;           // [16][16]
  float* rmax = red + 256;               // [16]
  float* rsum = rmax + 16;               // [16]
  float* PP   = rsum + 16;               // [4][16][16] PV split-K partials
  unsigned int*  masku = (unsigned int*)(PP + 4 * 256);  // [16][513] packed bytes
  const unsigned char* maskb = (const unsigned char*)masku; // row stride 2052 B

  const int tid  = threadIdx.x;
  const int wave = tid >> 5;             // 0..7
  const int lane = tid & 31;
  const int q0   = blockIdx.x * 16;
  const int bb   = blockIdx.y;

  // ---- one-time: zero AVG, stage mask slab (16 x 2048 int32 -> bytes) ----
  for (int idx = tid; idx < 16 * SEQ; idx += 256) {
    AVG[(idx >> 11) * SP + (idx & (SEQ - 1))] = 0.f;
  }
  for (int idx = tid; idx < 16 * (SEQ / 4); idx += 256) {
    const int r  = idx >> 9;            // row 0..15
    const int c4 = idx & 511;           // group of 4 cols
    const int4 m4 = *(const int4*)(mask + (((size_t)bb * SEQ + q0 + r) * SEQ) + c4 * 4);
    unsigned int packed = (m4.x != 0 ? 1u : 0u) | (m4.y != 0 ? 1u : 0u) << 8 |
                          (m4.z != 0 ? 1u : 0u) << 16 | (m4.w != 0 ? 1u : 0u) << 24;
    masku[r * 513 + c4] = packed;
  }
  __syncthreads();

  for (int h = 0; h < N_HEAD; ++h) {
    const size_t headLD = (size_t)(bb * N_HEAD + h);
    // --- scores: S = (Q*scale) K^T, mask applied from LDS bytes ---
    const _Float16* qbase = qh + (headLD * SEQ + q0) * DPH;
    const v16h aq0 = load_a(qbase, DPH, 0);
    const v16h aq1 = load_a(qbase, DPH, 32);
    for (int nt = wave; nt < SEQ / 16; nt += 8) {
      const int n0 = nt * 16;
      const _Float16* kbase = kh + (headLD * SEQ + n0) * DPH;   // B^T rows = k rows
      v8f acc = {0.f,0.f,0.f,0.f,0.f,0.f,0.f,0.f};
      acc = wmma16(aq0, load_bt(kbase, DPH, 0),  acc);
      acc = wmma16(aq1, load_bt(kbase, DPH, 32), acc);
      const int nl = lane & 15;
      const int mb = (lane >> 4) << 3;
#pragma unroll
      for (int i = 0; i < 8; ++i) {
        const int r   = mb + i;
        const int col = n0 + nl;
        S[r * SP + col] = (maskb[r * 2052 + col] != 0) ? -1e18f : acc[i];
      }
    }
    __syncthreads();

    // --- softmax over 2048 cols per row; 16 threads per row, 128 cols each ---
    const int row = tid >> 4;
    const int sub = tid & 15;
    const int c0  = sub * 128;
    float lmax = -3.4e38f;
    for (int cI = 0; cI < 128; ++cI) lmax = fmaxf(lmax, S[row * SP + c0 + cI]);
    red[row * 16 + sub] = lmax;
    __syncthreads();
    if (sub == 0) {
      float m = red[row * 16];
#pragma unroll
      for (int j = 1; j < 16; ++j) m = fmaxf(m, red[row * 16 + j]);
      rmax[row] = m;
    }
    __syncthreads();
    const float rm = rmax[row];
    float lsum = 0.f;
    for (int cI = 0; cI < 128; ++cI) {
      const float e = __expf(S[row * SP + c0 + cI] - rm);
      S[row * SP + c0 + cI] = e;
      lsum += e;
    }
    red[row * 16 + sub] = lsum;
    __syncthreads();
    if (sub == 0) {
      float t = 0.f;
#pragma unroll
      for (int j = 0; j < 16; ++j) t += red[row * 16 + j];
      rsum[row] = 1.f / t;
    }
    __syncthreads();
    const float inv = rsum[row];
    for (int cI = 0; cI < 128; ++cI) {
      const float p = S[row * SP + c0 + cI] * inv;
      S[row * SP + c0 + cI] = p;
      AVG[row * SP + c0 + cI] += p * (1.0f / N_HEAD);
    }
    __syncthreads();

    // --- PV: ctx_tile = probs(16x2048) @ V(2048x64).
    //     8 waves: (wave&3) -> 16-wide d tile, (wave>>2) -> half of K range.
    {
      const int dt    = wave & 3;
      const int d0    = dt * 16;
      const int khalf = wave >> 2;
      const int kbeg  = khalf * (SEQ / 2);
      const _Float16* vbase = vT + (headLD * DPH + d0) * SEQ;   // B^T rows = vT rows
      v8f acc = {0.f,0.f,0.f,0.f,0.f,0.f,0.f,0.f};
      for (int k0 = kbeg; k0 < kbeg + SEQ / 2; k0 += 32) {
        acc = wmma16(load_a_lds(S, SP, k0), load_bt(vbase, SEQ, k0), acc);
      }
      const int nl = lane & 15;
      const int mb = (lane >> 4) << 3;
      if (khalf == 1) {
#pragma unroll
        for (int i = 0; i < 8; ++i) PP[dt * 256 + (mb + i) * 16 + nl] = acc[i];
      }
      __syncthreads();
      if (khalf == 0) {
#pragma unroll
        for (int i = 0; i < 8; ++i) {
          const int l  = q0 + mb + i;
          const int cc = h * DPH + d0 + nl;
          const float v = acc[i] + PP[dt * 256 + (mb + i) * 16 + nl];
          ctx[((size_t)bb * SEQ + l) * D_MODEL + cc] = (_Float16)v;
        }
      }
    }
    __syncthreads();   // S reused next head
  }

  // --- write attn_avg * query_mask ---
  for (int idx = tid; idx < 16 * SEQ; idx += 256) {
    const int r   = idx >> 11;
    const int col = idx & (SEQ - 1);
    const float qm = qmask[(size_t)bb * SEQ + q0 + r];
    avg_out[((size_t)bb * SEQ + (q0 + r)) * SEQ + col] = AVG[r * SP + col] * qm;
  }
}

// ---------------------------------------------------------------------------
// Kernel 3: final projection.  out = (ctx @ fin_w^T + fin_b) * qm.  f32 out.
// grid = 4096, block = 256 (8 waves); wave -> one 16x16 tile of 512x64 tiles.
// ---------------------------------------------------------------------------
__global__ void mha_final_proj(const _Float16* __restrict__ ctx,   // [8192][1024]
                               const _Float16* __restrict__ fw,    // [1024][1024] = B^T
                               const float*    __restrict__ fb,
                               const float*    __restrict__ qmask,
                               float* __restrict__ out) {
  const int wave = threadIdx.x >> 5;
  const int lane = threadIdx.x & 31;
  const int tile = blockIdx.x * 8 + wave;
  const int m0   = (tile >> 6) * 16;
  const int n0   = (tile & 63) * 16;
  const _Float16* abase = ctx + (size_t)m0 * D_MODEL;
  const _Float16* bbase = fw  + (size_t)n0 * D_MODEL;

  v8f acc = {0.f,0.f,0.f,0.f,0.f,0.f,0.f,0.f};
  for (int k0 = 0; k0 < D_MODEL; k0 += 32) {
    acc = wmma16(load_a(abase, D_MODEL, k0), load_bt(bbase, D_MODEL, k0), acc);
  }

  const int nl = lane & 15;
  const int mb = (lane >> 4) << 3;
  const float bv = fb[n0 + nl];
#pragma unroll
  for (int i = 0; i < 8; ++i) {
    const int Mg = m0 + mb + i;
    out[(size_t)Mg * D_MODEL + n0 + nl] = (acc[i] + bv) * qmask[Mg];
  }
}

// ---------------------------------------------------------------------------
extern "C" void kernel_launch(void* const* d_in, const int* in_sizes, int n_in,
                              void* d_out, int out_size, void* d_ws, size_t ws_size,
                              hipStream_t stream) {
  const float* k_in  = (const float*)d_in[0];
  const float* v_in  = (const float*)d_in[1];
  const float* q_in  = (const float*)d_in[2];
  const int*   amask = (const int*)  d_in[3];
  const float* qmask = (const float*)d_in[4];
  const float* kqv_w = (const float*)d_in[5];
  const float* kqv_b = (const float*)d_in[6];
  const float* fin_w = (const float*)d_in[7];
  const float* fin_b = (const float*)d_in[8];

  _Float16* ws = (_Float16*)d_ws;
  size_t o = 0;
  _Float16* xk  = ws + o; o += (size_t)M_TOT * D_MODEL;       // k in f16
  _Float16* xq  = ws + o; o += (size_t)M_TOT * D_MODEL;       // q in f16
  _Float16* xv  = ws + o; o += (size_t)M_TOT * D_MODEL;       // v in f16
  _Float16* wh  = ws + o; o += (size_t)3 * D_MODEL * D_MODEL; // kqv_w f16
  _Float16* fwh = ws + o; o += (size_t)D_MODEL * D_MODEL;     // fin_w f16
  _Float16* qh  = ws + o; o += (size_t)M_TOT * D_MODEL;       // [B][H][L][64]
  _Float16* kh  = ws + o; o += (size_t)M_TOT * D_MODEL;       // [B][H][L][64]
  _Float16* vT  = ws + o; o += (size_t)M_TOT * D_MODEL;       // [B][H][64][L]
  _Float16* ctx = ws + o; o += (size_t)M_TOT * D_MODEL;       // [B*L][1024]

  float* out_ctx = (float*)d_out;
  float* out_avg = out_ctx + (size_t)M_TOT * D_MODEL;

  const int n1 = M_TOT * D_MODEL;
  mha_cvt_f16<<<(n1 + 255) / 256, 256, 0, stream>>>(k_in, xk, n1);
  mha_cvt_f16<<<(n1 + 255) / 256, 256, 0, stream>>>(q_in, xq, n1);
  mha_cvt_f16<<<(n1 + 255) / 256, 256, 0, stream>>>(v_in, xv, n1);
  const int n2 = 3 * D_MODEL * D_MODEL;
  mha_cvt_f16<<<(n2 + 255) / 256, 256, 0, stream>>>(kqv_w, wh, n2);
  const int n3 = D_MODEL * D_MODEL;
  mha_cvt_f16<<<(n3 + 255) / 256, 256, 0, stream>>>(fin_w, fwh, n3);

  mha_qkv_proj<<<dim3(M_TOT / 16, 48), 128, 0, stream>>>(xk, xq, xv, wh, kqv_b, qh, kh, vT);

  mha_attn<<<dim3(SEQ / 16, BATCH), 256, SMEM_BYTES, stream>>>(qh, kh, vT, amask, qmask,
                                                               ctx, out_avg);

  mha_final_proj<<<4096, 256, 0, stream>>>(ctx, fwh, fin_b, qmask, out_ctx);
}
